// HoughOnSwinMap_26053271617624
// MI455X (gfx1250) — matrix operand order repaired and checked by reading, not compiled, not run on hardware
//
#include <hip/hip_runtime.h>
#include <hip/hip_bf16.h>
#include <math.h>

// ---------------- problem constants ----------------
constexpr int Hc = 64, Wc = 64, Cc = 256, CW = 64, Pc = 4096;
constexpr int Tt = 60, Rr = 181, QB = Rr * Tt;   // 10860 hough bins
constexpr int Bb = 8;
constexpr int NT_Z = 679;                         // ceil(10860/16)

typedef __attribute__((ext_vector_type(16))) _Float16 v16h;
typedef __attribute__((ext_vector_type(8)))  _Float16 v8h;
typedef __attribute__((ext_vector_type(8)))  float    v8f;

__device__ __forceinline__ v8f wmma32(const v16h& a, const v16h& b, const v8f& c) {
  return __builtin_amdgcn_wmma_f32_16x16x32_f16(false, a, false, b, (short)0, c, false, false);
}

// A fragment: 16x32 f16 from row-major f16 matrix; per-lane 2x contiguous 16B -> b128 loads
__device__ __forceinline__ v16h load_a16(const _Float16* __restrict__ A, int lda,
                                         int m0, int k0, int lane) {
  const int mrow = m0 + (lane & 15);
  const int kh   = k0 + ((lane >> 4) << 4);      // +0 or +16
  const _Float16* row = A + (size_t)mrow * lda + kh;
  v8h lo = *(const v8h*)(row);                   // K: kh..kh+7
  v8h hi = *(const v8h*)(row + 8);               // K: kh+8..kh+15
  return __builtin_shufflevector(lo, hi, 0,1,2,3,4,5,6,7,8,9,10,11,12,13,14,15);
}

// B fragment: 32x16 from row-major fp32 matrix (ldb = N stride), cvt in flight
__device__ __forceinline__ v16h load_b_f32(const float* __restrict__ B, int ldb,
                                           int k0, int n0, int lane) {
  const int ncol = n0 + (lane & 15);
  const int kh   = k0 + ((lane >> 4) << 4);      // +0 or +16
  v16h b;
#pragma unroll
  for (int j = 0; j < 16; ++j)
    b[j] = (_Float16)B[(size_t)(kh + j) * ldb + ncol];
  return b;
}

// B fragment: 32x16 from row-major f16 matrix
__device__ __forceinline__ v16h load_b_f16(const _Float16* __restrict__ B, int ldb,
                                           int k0, int n0, int lane) {
  const int ncol = n0 + (lane & 15);
  const int kh   = k0 + ((lane >> 4) << 4);
  v16h b;
#pragma unroll
  for (int j = 0; j < 16; ++j)
    b[j] = B[(size_t)(kh + j) * ldb + ncol];
  return b;
}

// ---------------- stage 0: analytic Hough geometry (replaces 178MB vote matrix) -------
__global__ void k_ridx(unsigned short* __restrict__ ridx, int* __restrict__ cnt) {
  int p = blockIdx.x * blockDim.x + threadIdx.x;
  if (p >= Pc) return;
  double xs = (double)(p & 63) - 31.5;
  double ys = 31.5 - (double)(p >> 6);
  for (int t = 0; t < Tt; ++t) {
    double th  = (double)t * 3.0 * 0.017453292519943295;
    double rho = xs * cos(th) + ys * sin(th);
    int r = (int)rint(rho) + 90;                 // round-half-even == np.round
    int q = r * Tt + t;
    ridx[p * Tt + t] = (unsigned short)q;
    atomicAdd(&cnt[q], 1);
  }
}

__global__ void k_scan(const int* __restrict__ cnt, int* __restrict__ start) {
  if (blockIdx.x == 0 && threadIdx.x == 0) {
    int acc = 0;
    for (int q = 0; q < QB; ++q) { start[q] = acc; acc += cnt[q]; }
    start[QB] = acc;
  }
}

// deterministic CSR fill: per theta, pixels appended in ascending order
__global__ void k_fill(const unsigned short* __restrict__ ridx, const int* __restrict__ start,
                       unsigned short* __restrict__ pix) {
  __shared__ int cur[Rr];
  if (threadIdx.x != 0) return;
  int t = blockIdx.x;
  for (int r = 0; r < Rr; ++r) cur[r] = start[r * Tt + t];
  for (int p = 0; p < Pc; ++p) {
    int q = ridx[p * Tt + t];
    pix[cur[q / Tt]++] = (unsigned short)p;
  }
}

// ---------------- tiny: fp32 -> f16 copy (weights, converted once) --------------------
__global__ void k_cvt16(const float* __restrict__ src, _Float16* __restrict__ dst, int n) {
  int i = blockIdx.x * blockDim.x + threadIdx.x;
  if (i < n) dst[i] = (_Float16)src[i];
}

// ---------------- stage 1: y = W_to @ x + b_to  (M-strip: 4 wmma / k-step) ------------
__global__ void k_y(const float* __restrict__ x, const _Float16* __restrict__ Wto16,
                    const float* __restrict__ bto, float* __restrict__ y32,
                    _Float16* __restrict__ y16) {
  int wid  = (blockIdx.x * blockDim.x + threadIdx.x) >> 5;
  int lane = threadIdx.x & 31;
  int b  = wid >> 8;             // 256 ntiles per batch
  int nt = wid & 255;
  int n0 = nt * 16;
  const float* X = x + (size_t)b * Cc * Pc;
  v8f acc[4] = {};
  for (int k0 = 0; k0 < Cc; k0 += 32) {
    v16h bf = load_b_f32(X, Pc, k0, n0, lane);   // streamed operand: loaded once
#pragma unroll
    for (int mt = 0; mt < 4; ++mt) {
      v16h a = load_a16(Wto16, Cc, mt * 16, k0, lane);
      acc[mt] = wmma32(a, bf, acc[mt]);
    }
  }
  int ncol = n0 + (lane & 15);
  int mh   = (lane >> 4) << 3;
#pragma unroll
  for (int mt = 0; mt < 4; ++mt)
#pragma unroll
    for (int i = 0; i < 8; ++i) {
      int m = mt * 16 + mh + i;
      float v = acc[mt][i] + bto[m];
      size_t o = ((size_t)b * CW + m) * Pc + ncol;
      y32[o] = v;                 // fp32 path: Hough reduction
      y16[o] = (_Float16)v;       // f16 shadow: k_h2 WMMA operand (no extra rounding)
    }
}

// ---------------- stage 2: Hough gather  ht[b,c,q] = sum relu(y)/count ----------------
__global__ void k_ht(const float* __restrict__ y, const int* __restrict__ start,
                     const unsigned short* __restrict__ pix, _Float16* __restrict__ ht16) {
  int idx = blockIdx.x * blockDim.x + threadIdx.x;   // bc * QB + q, exact grid
  int q  = idx % QB;
  int bc = idx / QB;
  const float* ys = y + (size_t)bc * Pc;
  int s0 = start[q], s1 = start[q + 1];
  float s = 0.f;
  for (int i = s0; i < s1; ++i) s += fmaxf(ys[pix[i]], 0.f);
  int c = s1 - s0;
  ht16[(size_t)bc * QB + q] = (_Float16)(s / (float)(c > 0 ? c : 1));
}

// ---------------- stage 3: 9x1 rho-conv GEMM (M=64 strip, B gather reused 4x) ---------
__global__ void k_z(const _Float16* __restrict__ ht16, const _Float16* __restrict__ Wf16,
                    const float* __restrict__ bf_, float* __restrict__ z) {
  int wid  = (blockIdx.x * blockDim.x + threadIdx.x) >> 5;
  int lane = threadIdx.x & 31;
  int b  = wid / NT_Z;
  int nt = wid - b * NT_Z;
  int n0 = nt * 16;
  const _Float16* HT = ht16 + (size_t)b * CW * QB;
  int  q   = n0 + (lane & 15);
  bool qok = q < QB;
  int  r   = q / Tt;
  int  t   = q - r * Tt;
  v8f acc[4] = {};
  for (int k0 = 0; k0 < 576; k0 += 32) {
    v16h bfr;
    int kh = k0 + ((lane >> 4) << 4);
#pragma unroll
    for (int j = 0; j < 16; ++j) {               // im2col along rho, zero-padded
      int kk = kh + j;
      int ci = kk / 9, tap = kk - ci * 9;
      int rr = r + tap - 4;
      bfr[j] = (qok && rr >= 0 && rr < Rr) ? HT[(size_t)ci * QB + rr * Tt + t]
                                           : (_Float16)0.f;
    }
#pragma unroll
    for (int mt = 0; mt < 4; ++mt) {
      v16h a = load_a16(Wf16, 576, mt * 16, k0, lane);
      acc[mt] = wmma32(a, bfr, acc[mt]);
    }
  }
  if (q < QB) {
    int mh = (lane >> 4) << 3;
#pragma unroll
    for (int mt = 0; mt < 4; ++mt)
#pragma unroll
      for (int i = 0; i < 8; ++i) {
        int m = mt * 16 + mh + i;
        z[((size_t)b * CW + m) * QB + q] = acc[mt][i] + bf_[m];
      }
  }
}

// ---------------- shared: training-mode BN batch statistics per channel ---------------
__global__ void k_bnstats(const float* __restrict__ src, float* __restrict__ stats,
                          int nchan, int per, int nb) {
  int c = blockIdx.x;
  float s = 0.f, ss = 0.f;
  int N = nb * per;
  for (int idx = threadIdx.x; idx < N; idx += blockDim.x) {
    int b = idx / per, e = idx - b * per;
    float v = src[((size_t)b * nchan + c) * per + e];
    s += v; ss += v * v;
  }
  __shared__ float sh[256], sh2[256];
  sh[threadIdx.x] = s; sh2[threadIdx.x] = ss;
  __syncthreads();
  for (int o = 128; o > 0; o >>= 1) {
    if ((int)threadIdx.x < o) { sh[threadIdx.x] += sh[threadIdx.x + o]; sh2[threadIdx.x] += sh2[threadIdx.x + o]; }
    __syncthreads();
  }
  if (threadIdx.x == 0) {
    float mu  = sh[0] / (float)N;
    float var = sh2[0] / (float)N - mu * mu;
    stats[2 * c]     = mu;
    stats[2 * c + 1] = rsqrtf(var + 1e-5f);
  }
}

// ---------------- stage 4: inverse Hough: BN+ReLU on the fly, 60-way gather -----------
__global__ void k_iht(const float* __restrict__ z, const float* __restrict__ stats,
                      const float* __restrict__ gamma_f, const float* __restrict__ beta_f,
                      const unsigned short* __restrict__ ridx, _Float16* __restrict__ iht16) {
  int idx = blockIdx.x * blockDim.x + threadIdx.x;   // b*CW*Pc, exact grid
  int p = idx & (Pc - 1);
  int c = (idx >> 12) & (CW - 1);
  int b = idx >> 18;
  float rs = stats[2 * c + 1];
  float g  = gamma_f[c] * rs;
  float be = beta_f[c] - stats[2 * c] * g;           // BN(z) = g*z + be
  const float* Z = z + ((size_t)b * CW + c) * QB;
  const unsigned short* RX = ridx + p * Tt;
  float s = 0.f;
#pragma unroll 4
  for (int t = 0; t < Tt; ++t)
    s += fmaxf(g * Z[RX[t]] + be, 0.f);
  iht16[idx] = (_Float16)(s * (1.0f / 64.0f));       // / float(W)
}

// ---------------- stage 5: 3x3 conv over concat(y, iht): K=1152, gather reused 4x -----
__global__ void k_h2(const _Float16* __restrict__ y16, const _Float16* __restrict__ iht16,
                     const _Float16* __restrict__ Wcat16, const float* __restrict__ bcat,
                     _Float16* __restrict__ h216) {
  int wid  = (blockIdx.x * blockDim.x + threadIdx.x) >> 5;
  int lane = threadIdx.x & 31;
  int b  = wid >> 8;
  int nt = wid & 255;
  int n0 = nt * 16;
  const _Float16* Ys = y16   + (size_t)b * CW * Pc;
  const _Float16* Is = iht16 + (size_t)b * CW * Pc;
  int p  = n0 + (lane & 15);
  int py = p >> 6, px = p & 63;
  v8f acc[4] = {};
  for (int k0 = 0; k0 < 1152; k0 += 32) {
    v16h bfr;
    int kh = k0 + ((lane >> 4) << 4);
#pragma unroll
    for (int j = 0; j < 16; ++j) {               // im2col 3x3, zero-padded borders
      int kk  = kh + j;
      int ci  = kk / 9, tap = kk - ci * 9;
      int dy  = tap / 3 - 1, dx = tap - (tap / 3) * 3 - 1;
      int yy2 = py + dy, xx = px + dx;
      _Float16 v = (_Float16)0.f;
      if ((unsigned)yy2 < 64u && (unsigned)xx < 64u) {
        int pp = (yy2 << 6) + xx;
        v = (ci < CW) ? Ys[(size_t)ci * Pc + pp] : Is[(size_t)(ci - CW) * Pc + pp];
      }
      bfr[j] = v;
    }
#pragma unroll
    for (int mt = 0; mt < 4; ++mt) {
      v16h a = load_a16(Wcat16, 1152, mt * 16, k0, lane);
      acc[mt] = wmma32(a, bfr, acc[mt]);
    }
  }
  int ncol = n0 + (lane & 15);
  int mh   = (lane >> 4) << 3;
#pragma unroll
  for (int mt = 0; mt < 4; ++mt)
#pragma unroll
    for (int i = 0; i < 8; ++i) {
      int m = mt * 16 + mh + i;
      h216[((size_t)b * CW + m) * Pc + ncol] = (_Float16)(acc[mt][i] + bcat[m]);
    }
}

// ---------------- stage 6: conv1x1 to 256ch + residual (M=256 as 4x 64-strips) --------
__global__ void k_t(const _Float16* __restrict__ h216, const _Float16* __restrict__ Wfrom16,
                    const float* __restrict__ bfrom, const float* __restrict__ x,
                    float* __restrict__ out) {
  int wid  = (blockIdx.x * blockDim.x + threadIdx.x) >> 5;
  int lane = threadIdx.x & 31;
  int b  = wid >> 10;            // 4 mstrips * 256 ntiles
  int ms = (wid >> 8) & 3;
  int nt = wid & 255;
  int n0 = nt * 16;
  const _Float16* H = h216 + (size_t)b * CW * Pc;
  v8f acc[4] = {};
  for (int k0 = 0; k0 < CW; k0 += 32) {
    v16h bf = load_b_f16(H, Pc, k0, n0, lane);
#pragma unroll
    for (int mt = 0; mt < 4; ++mt) {
      v16h a = load_a16(Wfrom16, CW, ms * 64 + mt * 16, k0, lane);
      acc[mt] = wmma32(a, bf, acc[mt]);
    }
  }
  int ncol = n0 + (lane & 15);
  int mh   = (lane >> 4) << 3;
#pragma unroll
  for (int mt = 0; mt < 4; ++mt)
#pragma unroll
    for (int i = 0; i < 8; ++i) {
      int m = ms * 64 + mt * 16 + mh + i;
      size_t o = ((size_t)b * Cc + m) * Pc + ncol;
      out[o] = x[o] + acc[mt][i] + bfrom[m];     // ALPHA = 1.0
    }
}

// ---------------- stage 7: final BN applied in-place on d_out -------------------------
__global__ void k_bnapply(float* __restrict__ out, const float* __restrict__ stats,
                          const float* __restrict__ gamma, const float* __restrict__ beta) {
  int idx = blockIdx.x * blockDim.x + threadIdx.x;   // Bb*Cc*Pc, exact grid
  int c = (idx >> 12) & 255;
  out[idx] = gamma[c] * (out[idx] - stats[2 * c]) * stats[2 * c + 1] + beta[c];
}

// =====================================================================================
extern "C" void kernel_launch(void* const* d_in, const int* in_sizes, int n_in,
                              void* d_out, int out_size, void* d_ws, size_t ws_size,
                              hipStream_t stream) {
  (void)in_sizes; (void)n_in; (void)out_size; (void)ws_size;
  const float* x      = (const float*)d_in[0];
  const float* Wto    = (const float*)d_in[1];
  const float* bto    = (const float*)d_in[2];
  /* d_in[3] = vote (178 MB) intentionally unused: geometry recomputed analytically */
  const float* Wfilt  = (const float*)d_in[4];
  const float* bfilt  = (const float*)d_in[5];
  const float* gammaf = (const float*)d_in[6];
  const float* betaf  = (const float*)d_in[7];
  const float* Wcat   = (const float*)d_in[8];
  const float* bcat   = (const float*)d_in[9];
  const float* Wfrom  = (const float*)d_in[10];
  const float* bfrom  = (const float*)d_in[11];
  const float* gamma  = (const float*)d_in[12];
  const float* beta   = (const float*)d_in[13];
  float* out = (float*)d_out;

  // workspace carve-out (~56 MB total), 256B-aligned segments (all L2-resident)
  char* ws = (char*)d_ws;
  size_t off = 0;
  auto carve = [&](size_t bytes) -> char* {
    char* p = ws + off;
    off += (bytes + 255) & ~(size_t)255;
    return p;
  };
  unsigned short* ridx   = (unsigned short*)carve((size_t)Pc * Tt * 2);
  int*            cnt    = (int*)           carve((size_t)QB * 4);
  int*            start  = (int*)           carve((size_t)(QB + 1) * 4);
  unsigned short* pix    = (unsigned short*)carve((size_t)Pc * Tt * 2);
  _Float16*       wto16  = (_Float16*)      carve((size_t)CW * Cc * 2);
  _Float16*       wf16   = (_Float16*)      carve((size_t)CW * 576 * 2);
  _Float16*       wcat16 = (_Float16*)      carve((size_t)CW * 1152 * 2);
  _Float16*       wfrom16= (_Float16*)      carve((size_t)Cc * CW * 2);
  float*          y32    = (float*)         carve((size_t)Bb * CW * Pc * 4);   // 8.4 MB
  _Float16*       y16    = (_Float16*)      carve((size_t)Bb * CW * Pc * 2);   // 4.2 MB
  _Float16*       ht16   = (_Float16*)      carve((size_t)Bb * CW * QB * 2);   // 11.1 MB
  float*          zb     = (float*)         carve((size_t)Bb * CW * QB * 4);   // 22.2 MB
  float*          stf    = (float*)         carve((size_t)CW * 2 * 4);
  _Float16*       iht16  = (_Float16*)      carve((size_t)Bb * CW * Pc * 2);   // 4.2 MB
  _Float16*       h216   = (_Float16*)      carve((size_t)Bb * CW * Pc * 2);   // 4.2 MB
  float*          st2    = (float*)         carve((size_t)Cc * 2 * 4);

  hipMemsetAsync(cnt, 0, (size_t)QB * 4, stream);

  // geometry / CSR (deterministic ordering -> bit-identical replays)
  k_ridx<<<Pc / 256, 256, 0, stream>>>(ridx, cnt);
  k_scan<<<1, 32, 0, stream>>>(cnt, start);
  k_fill<<<Tt, 32, 0, stream>>>(ridx, start, pix);

  // one-time weight conversion to f16 (WMMA would round these identically per-MAC)
  k_cvt16<<<(CW * Cc   + 255) / 256, 256, 0, stream>>>(Wto,   wto16,   CW * Cc);
  k_cvt16<<<(CW * 576  + 255) / 256, 256, 0, stream>>>(Wfilt, wf16,    CW * 576);
  k_cvt16<<<(CW * 1152 + 255) / 256, 256, 0, stream>>>(Wcat,  wcat16,  CW * 1152);
  k_cvt16<<<(Cc * CW   + 255) / 256, 256, 0, stream>>>(Wfrom, wfrom16, Cc * CW);

  // main pipeline (each GEMM wave owns a full M-strip: 4 accumulators, B loaded once)
  k_y   <<<(Bb * 256) / 8,          256, 0, stream>>>(x, wto16, bto, y32, y16);
  k_ht  <<<(Bb * CW * QB) / 256,    256, 0, stream>>>(y32, start, pix, ht16);
  k_z   <<<(Bb * NT_Z) / 8,         256, 0, stream>>>(ht16, wf16, bfilt, zb);
  k_bnstats<<<CW, 256, 0, stream>>>(zb, stf, CW, QB, Bb);
  k_iht <<<(Bb * CW * Pc) / 256,    256, 0, stream>>>(zb, stf, gammaf, betaf, ridx, iht16);
  k_h2  <<<(Bb * 256) / 8,          256, 0, stream>>>(y16, iht16, wcat16, bcat, h216);
  k_t   <<<(Bb * 4 * 256) / 8,      256, 0, stream>>>(h216, wfrom16, bfrom, x, out);
  k_bnstats<<<Cc, 256, 0, stream>>>(out, st2, Cc, Pc, Bb);
  k_bnapply<<<(Bb * Cc * Pc) / 256, 256, 0, stream>>>(out, st2, gamma, beta);
}
